// GruDecoder_47579647705458
// MI455X (gfx1250) — matrix operand order repaired and checked by reading, not compiled, not run on hardware
//
#include <hip/hip_runtime.h>
#include <hip/hip_bf16.h>

#define MAXLEN  128
#define NSYM    32000
#define HID     1024
#define BATCH   32

typedef __attribute__((ext_vector_type(16))) __bf16 v16bf;
typedef __attribute__((ext_vector_type(8)))  __bf16 v8bf;
typedef __attribute__((ext_vector_type(8)))  float  v8f;
typedef unsigned short u16;
typedef unsigned long long u64;

// ---------- helpers ----------

__device__ __forceinline__ u16 f2bf(float f) {
    unsigned int u = __float_as_uint(f);
    u += 0x7FFFu + ((u >> 16) & 1u);          // round-to-nearest-even
    return (u16)(u >> 16);
}

__device__ __forceinline__ v8bf ldb8(const u16* p)  { return *(const v8bf*)p; }
__device__ __forceinline__ v16bf ldb16(const u16* p){ return *(const v16bf*)p; }

__device__ __forceinline__ v16bf cat16(v8bf lo, v8bf hi) {
    v16bf r;
#pragma unroll
    for (int i = 0; i < 8; ++i) { r[i] = lo[i]; r[i + 8] = hi[i]; }
    return r;
}

// A tile (16x32, bf16) from row-major [rows x ld]: lane = M, 16 contiguous K per lane half
__device__ __forceinline__ v16bf loadA(const u16* base, int m0, int k0, int ld, int lane) {
    const u16* row = base + (size_t)(m0 + (lane & 15)) * ld;
    return ldb16(row + k0 + ((lane >> 4) << 4));
}

// B tile (32x16, bf16): B[k][n] = act[n][k]; lane L -> act row n0+L%16, 16 contiguous K
__device__ __forceinline__ v16bf loadB(const u16* base, int n0, int k0, int ld, int lane) {
    const u16* row = base + (size_t)(n0 + (lane & 15)) * ld;
    return ldb16(row + k0 + ((lane >> 4) << 4));
}

__device__ __forceinline__ v8f wmma_bf(v16bf a, v16bf b, v8f c) {
    return __builtin_amdgcn_wmma_f32_16x16x32_bf16(false, a, false, b, (short)0, c, false, false);
}

__device__ __forceinline__ float sigmoidf(float x) { return 1.0f / (1.0f + __expf(-x)); }

// ---------- one-time setup kernels ----------

__global__ void k_convert(const float* __restrict__ wih, const float* __restrict__ whh,
                          const float* __restrict__ clsw,
                          u16* __restrict__ dwih, u16* __restrict__ dwhh, u16* __restrict__ dcls) {
    const long long NIH  = 3LL * HID * HID;
    const long long NCLS = (long long)NSYM * HID;
    const long long total = 2 * NIH + NCLS;
    const long long stride = (long long)gridDim.x * blockDim.x;
    for (long long i = (long long)blockIdx.x * blockDim.x + threadIdx.x; i < total; i += stride) {
        if (i < NIH)            dwih[i]           = f2bf(wih[i]);
        else if (i < 2 * NIH)   dwhh[i - NIH]     = f2bf(whh[i - NIH]);
        else                    dcls[i - 2 * NIH] = f2bf(clsw[i - 2 * NIH]);
    }
}

__global__ __launch_bounds__(256) void k_init_state(const float* __restrict__ h,
                                                    const float* __restrict__ emb,
                                                    float* __restrict__ hf0, u16* __restrict__ hbf0,
                                                    u16* __restrict__ xbf, u64* __restrict__ amax) {
    const int tid = blockIdx.x * 256 + threadIdx.x;      // 0 .. 32*1024-1
    const int k = tid & (HID - 1);
    hf0[tid]  = h[tid];
    hbf0[tid] = f2bf(h[tid]);
    xbf[tid]  = f2bf(emb[(size_t)NSYM * HID + k]);       // BOS row broadcast over batch
    if (tid < BATCH) amax[tid] = 0ull;
}

// ---------- per-step kernels ----------

// grid 64 blocks x 64 threads (2 waves); wave = batch half; block = 16 hidden units
__global__ __launch_bounds__(64) void k_gru(
    const u16* __restrict__ wih, const u16* __restrict__ whh,
    const float* __restrict__ bih, const float* __restrict__ bhh,
    const u16* __restrict__ xbf,
    const u16* __restrict__ hbf_cur, const float* __restrict__ hf_cur,
    u16* __restrict__ hbf_nxt, float* __restrict__ hf_nxt,
    u64* __restrict__ amax) {
    const int lane = threadIdx.x & 31;
    const int bt   = threadIdx.x >> 5;        // batch tile 0/1
    const int u0   = blockIdx.x * 16;

    if (blockIdx.x == 0 && threadIdx.x < BATCH) amax[threadIdx.x] = 0ull;  // reset for this step

    v8f ir = {}, iz = {}, in_ = {}, hr = {}, hz = {}, hn = {};
    for (int k0 = 0; k0 < HID; k0 += 32) {
        v16bf bx = loadB(xbf,     bt * 16, k0, HID, lane);
        v16bf bh = loadB(hbf_cur, bt * 16, k0, HID, lane);
        v16bf a;
        a = loadA(wih, 0 * HID + u0, k0, HID, lane); ir  = wmma_bf(a, bx, ir);
        a = loadA(wih, 1 * HID + u0, k0, HID, lane); iz  = wmma_bf(a, bx, iz);
        a = loadA(wih, 2 * HID + u0, k0, HID, lane); in_ = wmma_bf(a, bx, in_);
        a = loadA(whh, 0 * HID + u0, k0, HID, lane); hr  = wmma_bf(a, bh, hr);
        a = loadA(whh, 1 * HID + u0, k0, HID, lane); hz  = wmma_bf(a, bh, hz);
        a = loadA(whh, 2 * HID + u0, k0, HID, lane); hn  = wmma_bf(a, bh, hn);
    }
    const int n = (lane & 15) + 16 * bt;
#pragma unroll
    for (int r = 0; r < 8; ++r) {
        const int u = u0 + r + ((lane >> 4) << 3);
        const float rg = sigmoidf(ir[r] + bih[u]           + hr[r] + bhh[u]);
        const float zg = sigmoidf(iz[r] + bih[HID + u]     + hz[r] + bhh[HID + u]);
        const float ng = tanhf   (in_[r] + bih[2 * HID + u] + rg * (hn[r] + bhh[2 * HID + u]));
        const float hp = hf_cur[(size_t)n * HID + u];
        const float ho = (1.0f - zg) * ng + zg * hp;
        hf_nxt [(size_t)n * HID + u] = ho;
        hbf_nxt[(size_t)n * HID + u] = f2bf(ho);
    }
}

// grid 250 blocks x 128 threads (4 waves); each wave owns TWO adjacent 16-row vocab tiles
// and both batch halves: B registers reused across both A tiles -> 32 B/WMMA load traffic.
__global__ __launch_bounds__(128) void k_logits(
    const u16* __restrict__ clsw, const float* __restrict__ clsb,
    const u16* __restrict__ hbf,
    float* __restrict__ out,               // already offset to step t
    u64* __restrict__ amax) {
    __shared__ u64 red[4][32];
    const int lane = threadIdx.x & 31;
    const int wave = threadIdx.x >> 5;
    const int m0   = (blockIdx.x * 8 + wave * 2) * 16;   // first of two adjacent m-tiles

    v8f accs[2][2] = {{{}, {}}, {{}, {}}};               // [mtile][batch-tile]
    for (int k0 = 0; k0 < HID; k0 += 32) {
        v16bf b0 = loadB(hbf, 0,  k0, HID, lane);
        v16bf b1 = loadB(hbf, 16, k0, HID, lane);
        v16bf a0 = loadA(clsw, m0,      k0, HID, lane);
        v16bf a1 = loadA(clsw, m0 + 16, k0, HID, lane);
        accs[0][0] = wmma_bf(a0, b0, accs[0][0]);
        accs[0][1] = wmma_bf(a0, b1, accs[0][1]);
        accs[1][0] = wmma_bf(a1, b0, accs[1][0]);
        accs[1][1] = wmma_bf(a1, b1, accs[1][1]);
    }
    const int hi = lane >> 4;
    u64 best[2] = {0ull, 0ull};
#pragma unroll
    for (int mt = 0; mt < 2; ++mt) {
        const int mb = m0 + 16 * mt;
#pragma unroll
        for (int bt = 0; bt < 2; ++bt) {
            const int n = (lane & 15) + 16 * bt;
#pragma unroll
            for (int r = 0; r < 8; ++r) {
                const int m = mb + r + 8 * hi;
                const float v = accs[mt][bt][r] + clsb[m];
                out[(size_t)n * NSYM + m] = v;
                unsigned int ub = __float_as_uint(v);
                ub ^= (ub >> 31) ? 0xFFFFFFFFu : 0x80000000u;   // order-preserving map
                const u64 pk = ((u64)ub << 32) | (u64)(0xFFFFFFFFu - (unsigned)m);
                best[bt] = best[bt] > pk ? best[bt] : pk;
            }
        }
    }
    // lanes L and L+16 share column n but cover different rows: reduce across halves
#pragma unroll
    for (int bt = 0; bt < 2; ++bt) {
        unsigned int lo = (unsigned int)best[bt], hi32 = (unsigned int)(best[bt] >> 32);
        lo   = __shfl_xor(lo,   16, 32);
        hi32 = __shfl_xor(hi32, 16, 32);
        const u64 other = ((u64)hi32 << 32) | lo;
        best[bt] = best[bt] > other ? best[bt] : other;
    }
    if (lane < 16) { red[wave][lane] = best[0]; red[wave][lane + 16] = best[1]; }
    __syncthreads();
    if (wave == 0) {
        u64 b = red[0][lane];
#pragma unroll
        for (int w = 1; w < 4; ++w) { const u64 v = red[w][lane]; b = b > v ? b : v; }
        atomicMax(&amax[lane], b);
    }
}

// grid 128 blocks x 256 threads: x_{t+1} = bf16(emb[argmax])
__global__ __launch_bounds__(256) void k_gather(const u64* __restrict__ amax,
                                                const float* __restrict__ emb,
                                                u16* __restrict__ xbf) {
    const int tid = blockIdx.x * 256 + threadIdx.x;      // 0 .. 32*1024-1
    const int b = tid >> 10, k = tid & (HID - 1);
    const unsigned int tok = 0xFFFFFFFFu - (unsigned int)(amax[b] & 0xFFFFFFFFull);
    xbf[(size_t)b * HID + k] = f2bf(emb[(size_t)tok * HID + k]);
}

// ---------- host ----------

extern "C" void kernel_launch(void* const* d_in, const int* in_sizes, int n_in,
                              void* d_out, int out_size, void* d_ws, size_t ws_size,
                              hipStream_t stream) {
    (void)in_sizes; (void)n_in; (void)out_size; (void)ws_size;
    const float* h    = (const float*)d_in[0];
    const float* wih  = (const float*)d_in[1];
    const float* whh  = (const float*)d_in[2];
    const float* bih  = (const float*)d_in[3];
    const float* bhh  = (const float*)d_in[4];
    const float* clsw = (const float*)d_in[5];
    const float* clsb = (const float*)d_in[6];
    const float* emb  = (const float*)d_in[7];
    float* out = (float*)d_out;

    char* ws = (char*)d_ws;
    size_t off = 0;
    auto carve = [&](size_t bytes) -> char* {
        char* p = ws + off;
        off = (off + bytes + 255) & ~(size_t)255;
        return p;
    };
    u16*  dwih   = (u16*)carve(3ull * HID * HID * sizeof(u16));
    u16*  dwhh   = (u16*)carve(3ull * HID * HID * sizeof(u16));
    u16*  dcls   = (u16*)carve((size_t)NSYM * HID * sizeof(u16));
    float* hf[2]  = { (float*)carve((size_t)BATCH * HID * sizeof(float)),
                      (float*)carve((size_t)BATCH * HID * sizeof(float)) };
    u16*   hbf[2] = { (u16*)carve((size_t)BATCH * HID * sizeof(u16)),
                      (u16*)carve((size_t)BATCH * HID * sizeof(u16)) };
    u16*  xbf  = (u16*)carve((size_t)BATCH * HID * sizeof(u16));
    u64*  amax = (u64*)carve(BATCH * sizeof(u64));

    k_convert<<<2048, 256, 0, stream>>>(wih, whh, clsw, dwih, dwhh, dcls);
    k_init_state<<<(BATCH * HID) / 256, 256, 0, stream>>>(h, emb, hf[0], hbf[0], xbf, amax);

    for (int t = 0; t < MAXLEN; ++t) {
        const int cur = t & 1, nxt = cur ^ 1;
        k_gru<<<HID / 16, 64, 0, stream>>>(dwih, dwhh, bih, bhh, xbf,
                                           hbf[cur], hf[cur], hbf[nxt], hf[nxt], amax);
        k_logits<<<NSYM / (16 * 8), 128, 0, stream>>>(dcls, clsb, hbf[nxt],
                                                      out + (size_t)t * BATCH * NSYM, amax);
        k_gather<<<(BATCH * HID) / 256, 256, 0, stream>>>(amax, emb, xbf);
    }
}
